// E2E_27479200760299
// MI455X (gfx1250) — matrix operand order repaired and checked by reading, not compiled, not run on hardware
//
#include <hip/hip_runtime.h>
#include <hip/hip_bf16.h>
#include <math.h>

// ---------------------------------------------------------------------------
// Problem constants (reference): B=32 T=400 E=D=1024 V=5000 L=100
// ---------------------------------------------------------------------------
#define BB   32
#define TT   400
#define EE   1024
#define DD   1024
#define VV   5000
#define LL   100
#define L1   101
#define DE   2048
#define G4   4096
#define MROWS (L1*BB)      // 3232
#define LDV  5008          // padded logits stride
#define SOS_ID 4999
#define NEGI  -1e30f

typedef __attribute__((ext_vector_type(16))) __bf16 v16bf;
typedef __attribute__((ext_vector_type(8)))  float  v8f;

union Frag16 { v16bf v; unsigned int u[8]; uint4 q[2]; };
union FragC  { v8f   v; float        f[8]; };

// CDNA5 async global->LDS copy path (guarded; falls back to vector copies).
// Toolchain signature (probe-verified): (global int4*, local int4*, Ii, Ii)
#if __has_builtin(__builtin_amdgcn_global_load_async_to_lds_b128)
#define HAVE_ASYNC 1
typedef int async_v4i __attribute__((vector_size(16)));
typedef __attribute__((address_space(1))) async_v4i* gas_v4i_p;
typedef __attribute__((address_space(3))) async_v4i* las_v4i_p;
#define ASYNC_CP_B128(gsrc, ldst) \
    __builtin_amdgcn_global_load_async_to_lds_b128((gas_v4i_p)(gsrc), (las_v4i_p)(ldst), 0, 0)
#endif

#ifdef HAVE_ASYNC
# if __has_builtin(__builtin_amdgcn_s_wait_asynccnt)
#  define WAIT_ASYNC(n) __builtin_amdgcn_s_wait_asynccnt(n)
# else
#  define WAIT_ASYNC(n) asm volatile("s_wait_asynccnt " #n ::: "memory")
# endif
#else
# define WAIT_ASYNC(n) ((void)0)
#endif

__device__ __forceinline__ unsigned short f2bf(float x) {
    unsigned int u = __float_as_uint(x);
    u = (u + 0x7fffu + ((u >> 16) & 1u)) >> 16;
    return (unsigned short)u;
}
__device__ __forceinline__ float bf2f(unsigned short h) {
    return __uint_as_float(((unsigned int)h) << 16);
}
__device__ __forceinline__ float2 bfpair(unsigned int u) {
    float2 r;
    r.x = __uint_as_float(u << 16);
    r.y = __uint_as_float(u & 0xffff0000u);
    return r;
}
__device__ __forceinline__ float sigf(float x) { return 1.0f / (1.0f + expf(-x)); }

// ---------------------------------------------------------------------------
// Stage one 32(M)x32(K) A tile + 128(N)x32(K) B tile of bf16 into LDS.
// Pure 16-byte copies -> GLOBAL_LOAD_ASYNC_TO_LDS_B128 (ASYNCcnt-tracked)
// when the toolchain exposes it, uint4 copies otherwise. OOB B rows clamped
// (their results feed only columns the epilogue masks out).
// ---------------------------------------------------------------------------
__device__ __forceinline__ void stage_tile(
    const unsigned short* __restrict__ A, const unsigned short* __restrict__ Bw,
    unsigned short (*As)[40], unsigned short (*Bs)[40],
    int m0, int n0, int N, int K, int k0, int tid)
{
    const int seg = (tid & 3) * 8;          // 8 bf16 = 16 bytes
    const int ra  = tid >> 2;               // 0..31
    const unsigned short* ga = A + (size_t)(m0 + ra) * K + k0 + seg;
    unsigned short* la = &As[ra][seg];
#ifdef HAVE_ASYNC
    ASYNC_CP_B128(ga, la);
#else
    *(uint4*)la = *(const uint4*)ga;
#endif
    #pragma unroll
    for (int i = 0; i < 4; ++i) {
        int rb = (tid >> 2) + 32 * i;       // 0..127
        int nrow = n0 + rb; nrow = (nrow < N) ? nrow : (N - 1);
        const unsigned short* gb = Bw + (size_t)nrow * K + k0 + seg;
        unsigned short* lb = &Bs[rb][seg];
#ifdef HAVE_ASYNC
        ASYNC_CP_B128(gb, lb);
#else
        *(uint4*)lb = *(const uint4*)gb;
#endif
    }
#ifndef HAVE_ASYNC
    if (k0 + 32 < K) {
        __builtin_prefetch(ga + 32, 0, 1);
        __builtin_prefetch(Bw + (size_t)((n0 + (tid >> 2) < N) ? n0 + (tid >> 2) : N - 1) * K + k0 + 32, 0, 1);
    }
#endif
}

// ---------------------------------------------------------------------------
// Workhorse GEMM: C[m,n] = act( beta*C[m,n] + sum_k A[m,k]*Bw[n,k] + bias[n] )
// A: MxK bf16 row-major; Bw: NxK bf16 row-major (the W^T pattern everywhere).
// Block = 128 threads (4 waves); block tile 32x128; each wave owns 16x64 =
// 4 WMMA accumulators sharing one A fragment. Double-buffered LDS with async
// copies overlapping v_wmma_f32_16x16x32_bf16 compute. All fragments are
// loaded before the 4 WMMAs so the matrix pipe issues back-to-back.
// M multiple of 32 (12800 / 32 / 3232); K multiple of 32; N guarded.
// ---------------------------------------------------------------------------
__global__ void __launch_bounds__(128)
gemm_bf16_wmma(const unsigned short* __restrict__ A,
               const unsigned short* __restrict__ Bw,
               const float* __restrict__ bias,
               float* __restrict__ Cf, unsigned short* __restrict__ Cbf,
               int M, int N, int K, int ldc, int act, int beta)
{
    __shared__ unsigned short As[2][32][40];    //  5.0 KB (rows 80B -> 16B aligned)
    __shared__ unsigned short Bs[2][128][40];   // 20.0 KB

    const int tid  = threadIdx.x;
    const int lane = tid & 31;
    const int wave = tid >> 5;
    const int wm   = (wave & 1) * 16;           // wave M offset in block tile
    const int wn0  = (wave >> 1) * 64;          // wave N offset in block tile
    const int m0   = blockIdx.y * 32;
    const int n0   = blockIdx.x * 128;

    const int h    = lane >> 4;                 // half-wave
    const int mla  = lane & 15;                 // A row / B,C column
    const int klo  = h ? 8  : 0;                // A frag K bases (ISA 16-bit A 16x32)
    const int khi  = h ? 24 : 16;
    const int kbo  = h ? 8  : 0;                // B frag K base (ISA 16-bit B 32x16)

    FragC acc[4];
    #pragma unroll
    for (int j = 0; j < 4; ++j)
        #pragma unroll
        for (int r = 0; r < 8; ++r) acc[j].f[r] = 0.0f;

    const int nk = K >> 5;
    stage_tile(A, Bw, As[0], Bs[0], m0, n0, N, K, 0, tid);

    for (int kt = 0; kt < nk; ++kt) {
        const int cur = kt & 1;
        if (kt + 1 < nk) {
            stage_tile(A, Bw, As[cur ^ 1], Bs[cur ^ 1], m0, n0, N, K, (kt + 1) << 5, tid);
            WAIT_ASYNC(5);                       // first 5 (current tile) complete
        } else {
            WAIT_ASYNC(0);
        }
        __syncthreads();

        // load ALL fragments first (one dscnt wait), then burst 4 WMMAs
        Frag16 af, bf[4];
        af.q[0] = *(const uint4*)&As[cur][wm + mla][klo];
        af.q[1] = *(const uint4*)&As[cur][wm + mla][khi];
        #pragma unroll
        for (int j = 0; j < 4; ++j) {
            const unsigned short* bp = &Bs[cur][wn0 + 16 * j + mla][kbo];
            bf[j].q[0] = *(const uint4*)bp;
            bf[j].q[1] = *(const uint4*)(bp + 16);
        }
        #pragma unroll
        for (int j = 0; j < 4; ++j) {
            acc[j].v = __builtin_amdgcn_wmma_f32_16x16x32_bf16(
                           false, af.v, false, bf[j].v, (short)0, acc[j].v, false, false);
        }
        __syncthreads();
    }

    // epilogue: bias / beta accumulate / activation / fp32 and-or bf16 store
    #pragma unroll
    for (int j = 0; j < 4; ++j) {
        #pragma unroll
        for (int r = 0; r < 8; ++r) {
            int gm = m0 + wm + r + (h ? 8 : 0);
            int gn = n0 + wn0 + 16 * j + mla;
            if (gn < N) {
                size_t idx = (size_t)gm * ldc + gn;
                float val = acc[j].f[r];
                if (bias) val += bias[gn];
                if (beta && Cf) val += Cf[idx];
                if (act == 1) val = tanhf(val);
                if (Cf)  Cf[idx]  = val;
                if (Cbf) Cbf[idx] = f2bf(val);
            }
        }
    }
}

// fp32 -> bf16 one-time conversion (weights, hs_pad)
__global__ void __launch_bounds__(256)
cvt_bf16(unsigned short* __restrict__ dst, const float* __restrict__ src, int n)
{
    for (int i = blockIdx.x * 256 + threadIdx.x; i < n; i += gridDim.x * 256)
        dst[i] = f2bf(src[i]);
}

// ---------------------------------------------------------------------------
// Attention: per batch element. Scores vs bf16 pre_enc (uint4-unpacked),
// masked 2x scaling, softmax, context over hs_pad. Writes bf16 att_c and the
// z_sc[:,1024:] slot.
// ---------------------------------------------------------------------------
__global__ void __launch_bounds__(256)
attn_softmax_ctx(const unsigned short* __restrict__ preenc,
                 const unsigned short* __restrict__ dqbf,
                 const float* __restrict__ hs,
                 const int*   __restrict__ hlens,
                 unsigned short* __restrict__ attc,
                 unsigned short* __restrict__ zslot)
{
    __shared__ float dqs[DD];
    __shared__ float ws[TT];
    __shared__ float red[256];
    const int b = blockIdx.x, tid = threadIdx.x;

    for (int d = tid; d < DD; d += 256) dqs[d] = bf2f(dqbf[b * DD + d]);
    __syncthreads();

    const int hl = hlens[b];
    float lmax = -3.4e38f;
    for (int t = tid; t < TT; t += 256) {
        const uint4* row = (const uint4*)(preenc + (size_t)(b * TT + t) * DD);
        float a = 0.0f;
        #pragma unroll 4
        for (int q = 0; q < DD / 8; ++q) {
            uint4 u = row[q];
            float2 p0 = bfpair(u.x), p1 = bfpair(u.y);
            float2 p2 = bfpair(u.z), p3 = bfpair(u.w);
            const float* dv = &dqs[q * 8];
            a += p0.x * dv[0] + p0.y * dv[1] + p1.x * dv[2] + p1.y * dv[3]
               + p2.x * dv[4] + p2.y * dv[5] + p3.x * dv[6] + p3.y * dv[7];
        }
        float e = (t < hl) ? 2.0f * a : NEGI;
        ws[t] = e;
        lmax  = fmaxf(lmax, e);
    }
    red[tid] = lmax; __syncthreads();
    for (int s = 128; s > 0; s >>= 1) { if (tid < s) red[tid] = fmaxf(red[tid], red[tid + s]); __syncthreads(); }
    float mx = red[0]; __syncthreads();

    float lsum = 0.0f;
    for (int t = tid; t < TT; t += 256) { float v = expf(ws[t] - mx); ws[t] = v; lsum += v; }
    red[tid] = lsum; __syncthreads();
    for (int s = 128; s > 0; s >>= 1) { if (tid < s) red[tid] += red[tid + s]; __syncthreads(); }
    float inv = 1.0f / red[0]; __syncthreads();

    for (int c = tid; c < EE; c += 256) {
        float a = 0.0f;
        #pragma unroll 4
        for (int t = 0; t < TT; ++t) a += ws[t] * hs[(size_t)(b * TT + t) * EE + c];
        unsigned short v = f2bf(a * inv);
        attc [b * EE + c]      = v;
        zslot[b * DE + DD + c] = v;              // z_sc = [z1, att_c]
    }
}

// x = [embed[ys_in[:,l]], att_c]  (bf16, GEMM A operand)
__global__ void __launch_bounds__(256)
build_x(const float* __restrict__ embed, const int* __restrict__ ys,
        const unsigned short* __restrict__ attc, unsigned short* __restrict__ x, int l)
{
    int idx = blockIdx.x * 256 + threadIdx.x;   // 32*1024 threads
    int b = idx >> 10, d = idx & 1023;
    int tok = (l == 0) ? SOS_ID : ys[b * LL + (l - 1)];
    x[b * DE + d]      = f2bf(embed[(size_t)tok * DD + d]);
    x[b * DE + DD + d] = attc[b * EE + d];
}

// LSTM gate nonlinearity + state update (gate order i,f,g,o)
__global__ void __launch_bounds__(256)
lstm_act(const float* __restrict__ gates, unsigned short* __restrict__ hbf,
         float* __restrict__ c, unsigned short* __restrict__ zslot)
{
    int idx = blockIdx.x * 256 + threadIdx.x;   // 32*1024 threads
    int b = idx >> 10, d = idx & 1023;
    const float* g = gates + (size_t)b * G4;
    float gi = g[d], gf = g[DD + d], gg = g[2 * DD + d], go = g[3 * DD + d];
    float cn = sigf(gf) * c[idx] + sigf(gi) * tanhf(gg);
    unsigned short hn = f2bf(sigf(go) * tanhf(cn));
    c[idx]   = cn;
    hbf[idx] = hn;
    if (zslot) zslot[b * DE + d] = hn;          // z_sc = [z1, att_c]
}

// Per-row cross entropy from logits
__global__ void __launch_bounds__(256)
ce_rows(const float* __restrict__ logits, const int* __restrict__ ys,
        float* __restrict__ ce)
{
    __shared__ float red[256];
    const int r = blockIdx.x, tid = threadIdx.x;
    const float* row = logits + (size_t)r * LDV;

    float lmax = -3.4e38f;
    for (int v = tid; v < VV; v += 256) lmax = fmaxf(lmax, row[v]);
    red[tid] = lmax; __syncthreads();
    for (int s = 128; s > 0; s >>= 1) { if (tid < s) red[tid] = fmaxf(red[tid], red[tid + s]); __syncthreads(); }
    float mx = red[0]; __syncthreads();

    float lsum = 0.0f;
    for (int v = tid; v < VV; v += 256) lsum += expf(row[v] - mx);
    red[tid] = lsum; __syncthreads();
    for (int s = 128; s > 0; s >>= 1) { if (tid < s) red[tid] += red[tid + s]; __syncthreads(); }

    if (tid == 0) {
        int l = r >> 5, b = r & 31;             // r = l*32 + b
        int tgt = (l < LL) ? ys[b * LL + l] : SOS_ID;
        ce[r] = mx + logf(red[0]) - row[tgt];
    }
}

__global__ void __launch_bounds__(256)
loss_reduce(const float* __restrict__ ce, float* __restrict__ out)
{
    __shared__ float red[256];
    int tid = threadIdx.x;
    float s = 0.0f;
    for (int r = tid; r < MROWS; r += 256) s += ce[r];
    red[tid] = s; __syncthreads();
    for (int st = 128; st > 0; st >>= 1) { if (tid < st) red[tid] += red[tid + st]; __syncthreads(); }
    if (tid == 0) out[0] = red[0] * ((float)LL / (float)MROWS);
}

__global__ void fill0_u32(unsigned int* p, int n)
{
    int i = blockIdx.x * 256 + threadIdx.x;
    if (i < n) p[i] = 0u;
}

// ---------------------------------------------------------------------------
extern "C" void kernel_launch(void* const* d_in, const int* in_sizes, int n_in,
                              void* d_out, int out_size, void* d_ws, size_t ws_size,
                              hipStream_t stream)
{
    (void)in_sizes; (void)n_in; (void)out_size; (void)ws_size;
    const float* hs    = (const float*)d_in[0];
    const float* embed = (const float*)d_in[1];
    const float* Wenc  = (const float*)d_in[2];
    const float* benc  = (const float*)d_in[3];
    const float* Wdec  = (const float*)d_in[4];
    const float* bdec  = (const float*)d_in[5];
    const float* Wih0  = (const float*)d_in[6];
    const float* Whh0  = (const float*)d_in[7];
    const float* bih0  = (const float*)d_in[8];
    const float* bhh0  = (const float*)d_in[9];
    const float* Wih1  = (const float*)d_in[10];
    const float* Whh1  = (const float*)d_in[11];
    const float* bih1  = (const float*)d_in[12];
    const float* bhh1  = (const float*)d_in[13];
    const float* Wout  = (const float*)d_in[14];
    const float* bout  = (const float*)d_in[15];
    const int*   hlens = (const int*)d_in[16];
    const int*   ys    = (const int*)d_in[17];

    // ---- workspace carve ----
    char* base = (char*)d_ws; size_t off = 0;
    auto carve = [&](size_t bytes) -> void* {
        void* p = base + off; off += bytes; off = (off + 255) & ~(size_t)255; return p;
    };
    typedef unsigned short us;
    us* hsbf    = (us*)carve((size_t)BB*TT*EE*2);
    us* preenc  = (us*)carve((size_t)BB*TT*DD*2);
    us* wencb   = (us*)carve((size_t)DD*EE*2);
    us* wdecb   = (us*)carve((size_t)DD*DD*2);
    us* wih0b   = (us*)carve((size_t)G4*DE*2);
    us* whh0b   = (us*)carve((size_t)G4*DD*2);
    us* wih1b   = (us*)carve((size_t)G4*DD*2);
    us* whh1b   = (us*)carve((size_t)G4*DD*2);
    us* woutb   = (us*)carve((size_t)VV*DE*2);
    us* zall    = (us*)carve((size_t)L1*BB*DE*2);
    float* logits = (float*)carve((size_t)MROWS*LDV*4);
    us* zbf     = (us*)carve((size_t)2*BB*DD*2);       // z0bf, z1bf
    float* cst  = (float*)carve((size_t)2*BB*DD*4);    // c0, c1
    us* z0bf = zbf,        *z1bf = zbf + BB*DD;
    float* c0 = cst,       *c1 = cst + BB*DD;
    us* dqbf    = (us*)carve((size_t)BB*DD*2);
    us* attcbf  = (us*)carve((size_t)BB*EE*2);
    us* xbf     = (us*)carve((size_t)BB*DE*2);
    float* gates = (float*)carve((size_t)BB*G4*4);
    float* ce   = (float*)carve((size_t)MROWS*4);

    // ---- one-time fp32 -> bf16 conversions (weights stay bf16 in L2) ----
    cvt_bf16<<<2048, 256, 0, stream>>>(hsbf,  hs,   BB*TT*EE);
    cvt_bf16<<<2048, 256, 0, stream>>>(wencb, Wenc, DD*EE);
    cvt_bf16<<<2048, 256, 0, stream>>>(wdecb, Wdec, DD*DD);
    cvt_bf16<<<2048, 256, 0, stream>>>(wih0b, Wih0, G4*DE);
    cvt_bf16<<<2048, 256, 0, stream>>>(whh0b, Whh0, G4*DD);
    cvt_bf16<<<2048, 256, 0, stream>>>(wih1b, Wih1, G4*DD);
    cvt_bf16<<<2048, 256, 0, stream>>>(whh1b, Whh1, G4*DD);
    cvt_bf16<<<2048, 256, 0, stream>>>(woutb, Wout, VV*DE);

    fill0_u32<<<(BB*DD + 255)/256, 256, 0, stream>>>((unsigned int*)zbf, BB*DD);       // both z bf16
    fill0_u32<<<(2*BB*DD + 255)/256, 256, 0, stream>>>((unsigned int*)cst, 2*BB*DD);   // both c fp32

    // ---- encoder precompute: pre_enc = tanh(hs @ Wenc^T + benc) -> bf16 ----
    gemm_bf16_wmma<<<dim3(DD/128, (BB*TT)/32), 128, 0, stream>>>(
        hsbf, wencb, benc, nullptr, preenc, BB*TT, DD, EE, DD, /*tanh*/1, 0);

    // ---- sequential decode ----
    for (int l = 0; l < L1; ++l) {
        us* zslot = zall + (size_t)l * BB * DE;
        // dq = tanh(z0 @ Wdec^T + bdec)
        gemm_bf16_wmma<<<dim3(DD/128, 1), 128, 0, stream>>>(
            z0bf, wdecb, bdec, nullptr, dqbf, BB, DD, DD, DD, 1, 0);
        attn_softmax_ctx<<<BB, 256, 0, stream>>>(preenc, dqbf, hs, hlens, attcbf, zslot);
        build_x<<<(BB*DD)/256, 256, 0, stream>>>(embed, ys, attcbf, xbf, l);
        // layer 0: gates = x@Wih0^T + bih0 + z0@Whh0^T + bhh0
        gemm_bf16_wmma<<<dim3(G4/128, 1), 128, 0, stream>>>(
            xbf,  wih0b, bih0, gates, nullptr, BB, G4, DE, G4, 0, 0);
        gemm_bf16_wmma<<<dim3(G4/128, 1), 128, 0, stream>>>(
            z0bf, whh0b, bhh0, gates, nullptr, BB, G4, DD, G4, 0, 1);
        lstm_act<<<(BB*DD)/256, 256, 0, stream>>>(gates, z0bf, c0, nullptr);
        // layer 1: gates = z0@Wih1^T + bih1 + z1@Whh1^T + bhh1
        gemm_bf16_wmma<<<dim3(G4/128, 1), 128, 0, stream>>>(
            z0bf, wih1b, bih1, gates, nullptr, BB, G4, DD, G4, 0, 0);
        gemm_bf16_wmma<<<dim3(G4/128, 1), 128, 0, stream>>>(
            z1bf, whh1b, bhh1, gates, nullptr, BB, G4, DD, G4, 0, 1);
        lstm_act<<<(BB*DD)/256, 256, 0, stream>>>(gates, z1bf, c1, zslot);
    }

    // ---- output projection + CE ----
    gemm_bf16_wmma<<<dim3((VV + 127)/128, MROWS/32), 128, 0, stream>>>(
        zall, woutb, bout, logits, nullptr, MROWS, VV, DE, LDV, 0, 0);
    ce_rows<<<MROWS, 256, 0, stream>>>(logits, ys, ce);
    loss_reduce<<<1, 256, 0, stream>>>(ce, (float*)d_out);
}
